// GlobalLocalMultiHeadAttention_28845000360708
// MI455X (gfx1250) — compile-verified, matrix operand order checked by reading, not run on hardware
//
#include <hip/hip_runtime.h>
#include <hip/hip_bf16.h>

// ---- CDNA5 WMMA / TDM types -------------------------------------------------
typedef __attribute__((ext_vector_type(16))) __bf16       v16bf;
typedef __attribute__((ext_vector_type(8)))  float        v8f;
typedef __attribute__((ext_vector_type(4)))  unsigned int u32x4;
typedef __attribute__((ext_vector_type(8)))  int          i32x8;
typedef __attribute__((ext_vector_type(4)))  int          i32x4;

// bf16 <-> f32 via bit manipulation (RNE); storage type = unsigned short.
__device__ __forceinline__ unsigned short f2bf(float f) {
    unsigned int u = __builtin_bit_cast(unsigned int, f);
    u += 0x7fffu + ((u >> 16) & 1u);
    return (unsigned short)(u >> 16);
}

// ISA 7.12.2 16-bit A/B fragment K-swizzle: slot e (0..15), lane-half lhi:
//   K = (e>>3)*16 + (e&7) + 8*lhi
__device__ __forceinline__ void fragpos(int K, int& lh16, int& e) {
    lh16 = (K & 8) ? 16 : 0;
    e    = (K & 7) + ((K >> 4) << 3);
}

#define B_   8
#define N_   2048
#define L_   64
#define DS_  1024
#define DR_  768
#define H_   8
#define HD_  96

// ---- Tensor Data Mover: 1-D 24576-byte tile, global -> LDS ------------------
// D# built per CDNA5 ISA: count=1, type=2(image), data_size=4B,
// tensor_dim0 = tile_dim0 = 6144 elems, dim0_stride = 6144, no pad/iterate.
__device__ __forceinline__ void tdm_load_b24k(const unsigned short* src,
                                              unsigned int ldsAddr) {
    unsigned long long ga = (unsigned long long)(size_t)src;
    u32x4 g0;
    g0[0] = 1u;                                            // count=1, user mode
    g0[1] = ldsAddr;                                       // lds_addr
    g0[2] = (unsigned int)ga;                              // global_addr[31:0]
    g0[3] = (unsigned int)((ga >> 32) & 0x01ffffffu) | 0x80000000u; // [56:32]+type=2
    i32x8 g1;
    g1[0] = 0x20000;      // data_size = 2 (4 bytes)
    g1[1] = 0x18000000;   // tensor_dim0[15:0]=6144 (bits 79:48 low half)
    g1[2] = 0x10000;      // tensor_dim0 hi=0, tensor_dim1=1
    g1[3] = 0x18000000;   // tile_dim0 = 6144 (bits 127:112)
    g1[4] = 0;            // tile_dim1=0 (unused), tile_dim2=0
    g1[5] = 6144;         // tensor_dim0_stride lo
    g1[6] = 0; g1[7] = 0;
    i32x4 gz = {0, 0, 0, 0};
#if defined(__clang_major__) && (__clang_major__ >= 23)
    i32x8 gz8 = {0, 0, 0, 0, 0, 0, 0, 0};
    __builtin_amdgcn_tensor_load_to_lds(g0, g1, gz, gz, gz8, 0);
#else
    __builtin_amdgcn_tensor_load_to_lds(g0, g1, gz, gz, 0);
#endif
}

// ============================================================================
// Kernel 0: Wq -> bf16, fragment order: [pass2][kchunk32][tile24][lane32][e16]
// 3 MB read once; output 1.5 MB stays L2-resident for all 256 fused blocks.
// ============================================================================
__global__ __launch_bounds__(256, 1)
void wqprep_kernel(const float* __restrict__ Wq, unsigned short* __restrict__ Wqf) {
    const int blk  = blockIdx.x;          // pass*32 + ck
    const int pass = blk >> 5, ck = blk & 31;
    const int c0 = pass * 384, kc = ck * 32;
    for (int idx = threadIdx.x; idx < 12288; idx += 256) {
        int T = idx >> 9, rem = idx & 511;
        int lanee = rem >> 4, e = rem & 15;
        int n = lanee & 15, lh = lanee >> 4;
        int K = ((e >> 3) << 4) + (e & 7) + 8 * lh;
        Wqf[(size_t)blk * 12288 + idx] =
            f2bf(Wq[(size_t)(kc + K) * DR_ + c0 + T * 16 + n]);
    }
}

// ============================================================================
// Kernel 1: K/V = LN(u)@{Wk,Wv}+bias -> bf16 workspace in per-head FRAGMENT
// order (K: alpha B-frags [tc4][ks3][32][16]; V: AV B-frags [ks2][tc6][32][16])
// so the fused kernel stages them with pure b128 copies. grid=8, block=256.
// ============================================================================
__global__ __launch_bounds__(256, 1)
void kv_kernel(const float* __restrict__ u,
               const float* __restrict__ gk, const float* __restrict__ bk,
               const float* __restrict__ gv, const float* __restrict__ bv,
               const float* __restrict__ Wk, const float* __restrict__ bkl,
               const float* __restrict__ Wv, const float* __restrict__ bvl,
               unsigned short* __restrict__ Kbf, unsigned short* __restrict__ Vbf) {
    __shared__ float sMu[L_], sRstd[L_];
    __shared__ float sU[64 * 64];
    const int b    = blockIdx.x;
    const int tid  = threadIdx.x;
    const int lane = tid & 31;
    const int w    = tid >> 5;

    for (int i = 0; i < 8; ++i) {
        int row = w * 8 + i;
        const float* ur = u + (size_t)(b * L_ + row) * DR_;
        float s = 0.f, s2 = 0.f;
        for (int t = 0; t < 24; ++t) {
            float v = ur[lane + 32 * t];
            s += v; s2 += v * v;
        }
        for (int o = 16; o > 0; o >>= 1) {
            s  += __shfl_xor(s,  o, 32);
            s2 += __shfl_xor(s2, o, 32);
        }
        if (lane == 0) {
            float mu  = s * (1.f / DR_);
            float var = s2 * (1.f / DR_) - mu * mu;
            sMu[row]   = mu;
            sRstd[row] = rsqrtf(var + 1e-5f);
        }
    }
    __syncthreads();

    for (int ct = 0; ct < 3; ++ct) {
        const int jcol = ct * 256 + tid;
        float accK[64], accV[64];
#pragma unroll
        for (int r = 0; r < 64; ++r) { accK[r] = 0.f; accV[r] = 0.f; }
        float bK = bkl[jcol], bV = bvl[jcol];

        for (int dc = 0; dc < DR_; dc += 64) {
            __syncthreads();
            for (int t = 0; t < 16; ++t) {
                int idx = tid + 256 * t;
                int row = idx >> 6, dd = idx & 63;
                float v = u[(size_t)(b * L_ + row) * DR_ + dc + dd];
                sU[idx] = (v - sMu[row]) * sRstd[row];
            }
            __syncthreads();
            for (int dd = 0; dd < 64; ++dd) {
                int d = dc + dd;
                float wkv = Wk[(size_t)d * DR_ + jcol];
                float wvv = Wv[(size_t)d * DR_ + jcol];
                float ak  = gk[d] * wkv;
                float av  = gv[d] * wvv;
                bK += bk[d] * wkv;
                bV += bv[d] * wvv;
#pragma unroll
                for (int r = 0; r < 64; ++r) {
                    float nu = sU[r * 64 + dd];
                    accK[r] = fmaf(nu, ak, accK[r]);
                    accV[r] = fmaf(nu, av, accV[r]);
                }
            }
        }
        // frag-order scatter (tiny kernel, off critical path)
        const int j = jcol / HD_, d = jcol - j * HD_;
        unsigned short* Kh = Kbf + (size_t)(b * H_ + j) * 6144;
        unsigned short* Vh = Vbf + (size_t)(b * H_ + j) * 6144;
        int ksd = d >> 5, kkd = d & 31, lhK, eK; fragpos(kkd, lhK, eK);
        int tcv = d >> 4, nv = d & 15;
#pragma unroll
        for (int r = 0; r < 64; ++r) {
            int tc = r >> 4, n = r & 15;
            Kh[((tc * 3 + ksd) * 32 + n + lhK) * 16 + eK] = f2bf(accK[r] + bK);
            int ksl = r >> 5, kkl = r & 31, lhV, eV; fragpos(kkl, lhV, eV);
            Vh[((ksl * 6 + tcv) * 32 + nv + lhV) * 16 + eV] = f2bf(accV[r] + bV);
        }
    }
}

// ============================================================================
// Kernel 2: k_beta = u_flat @ Wkb + bkb. Streams 192 MB of Wkb exactly once
// (the HBM floor of the whole problem), with global_prefetch_b8 lookahead.
// ============================================================================
__global__ __launch_bounds__(256, 1)
void kbeta_kernel(const float* __restrict__ u, const float* __restrict__ Wkb,
                  const float* __restrict__ bkb, float* __restrict__ kbeta) {
    const int tid = threadIdx.x;
    const int jl  = tid & 63;
    const int j   = blockIdx.x * 64 + jl;
    const int kg  = tid >> 6;
    const int KTOT = L_ * DR_;
    float acc[B_];
#pragma unroll
    for (int bb = 0; bb < B_; ++bb) acc[bb] = 0.f;

    for (int k = kg; k < KTOT; k += 4) {
        float wv = Wkb[(size_t)k * DS_ + j];
        __builtin_prefetch(&Wkb[(size_t)(k + 16) * DS_ + j], 0, 1);
#pragma unroll
        for (int bb = 0; bb < B_; ++bb)
            acc[bb] = fmaf(u[(size_t)bb * KTOT + k], wv, acc[bb]);
    }

    __shared__ float sred[4 * 64 * B_];
#pragma unroll
    for (int bb = 0; bb < B_; ++bb) sred[(kg * 64 + jl) * B_ + bb] = acc[bb];
    __syncthreads();
    if (tid < 64) {
        for (int bb = 0; bb < B_; ++bb) {
            float s = sred[(0 * 64 + tid) * B_ + bb] + sred[(1 * 64 + tid) * B_ + bb] +
                      sred[(2 * 64 + tid) * B_ + bb] + sred[(3 * 64 + tid) * B_ + bb];
            kbeta[bb * DS_ + blockIdx.x * 64 + tid] = s + bkb[blockIdx.x * 64 + tid];
        }
    }
}

// ============================================================================
// Kernel 3: w_b[b,:] = Wqb @ k_beta[b,:];  c_b[b] = bqb . k_beta[b]
// Folds the 34-GFLOP q_beta GEMM into a 16-MFLOP mat-vec.
// ============================================================================
__global__ __launch_bounds__(256, 1)
void wb_kernel(const float* __restrict__ Wqb, const float* __restrict__ bqb,
               const float* __restrict__ kbeta, float* __restrict__ wbv,
               float* __restrict__ cb) {
    __shared__ float sred[256 * B_];
    const int tid = threadIdx.x;
    const int blk = blockIdx.x;
    float acc[B_];
#pragma unroll
    for (int bb = 0; bb < B_; ++bb) acc[bb] = 0.f;

    if (blk < DS_) {
        const float* row = Wqb + (size_t)blk * DS_;
        for (int d = tid; d < DS_; d += 256) {
            float v = row[d];
#pragma unroll
            for (int bb = 0; bb < B_; ++bb)
                acc[bb] = fmaf(v, kbeta[bb * DS_ + d], acc[bb]);
        }
    } else {
        int bb0 = blk - DS_;
        for (int d = tid; d < DS_; d += 256)
            acc[0] = fmaf(bqb[d], kbeta[bb0 * DS_ + d], acc[0]);
    }
#pragma unroll
    for (int bb = 0; bb < B_; ++bb) sred[bb * 256 + tid] = acc[bb];
    __syncthreads();
    for (int s = 128; s > 0; s >>= 1) {
        if (tid < s) {
#pragma unroll
            for (int bb = 0; bb < B_; ++bb)
                sred[bb * 256 + tid] += sred[bb * 256 + tid + s];
        }
        __syncthreads();
    }
    if (blk < DS_) {
        if (tid < B_) wbv[tid * DS_ + blk] = sred[tid * 256];
    } else {
        if (tid == 0) cb[blk - DS_] = sred[0];
    }
}

// ============================================================================
// Kernel 4 (fused): 64 rows of p per block. LN+beta; q-GEMM with TDM-fed,
// double-buffered LDS (tensor_load_to_lds for chunk k+1 overlaps WMMA of
// chunk k, drained via s_wait_tensorcnt); per-head WMMA attention.
// LDS: 2 x (4KB A + 24KB B) ping-pong = 56KB (GEMM) / 52KB (attention).
// ============================================================================
__global__ __launch_bounds__(256, 1)
void fused_kernel(const float* __restrict__ p,
                  const float* __restrict__ gq, const float* __restrict__ bq,
                  const unsigned short* __restrict__ Wqf,
                  const float* __restrict__ bql,
                  const unsigned short* __restrict__ Kbf,
                  const unsigned short* __restrict__ Vbf,
                  const float* __restrict__ wbv, const float* __restrict__ cb,
                  float* __restrict__ out) {
    __shared__ float sMu[64], sRstd[64], sBeta[64];
    __shared__ __align__(64) unsigned char sBuf[57344];

    const int b    = blockIdx.x >> 5;
    const int n0   = (blockIdx.x & 31) * 64;
    const int tid  = threadIdx.x;
    const int lane = tid & 31;
    const int w    = tid >> 5;
    const int lhi  = (lane >= 16);
    const int m    = lane & 15;

    // ---- stats / beta / passthrough ----
    {
        float* sWb = (float*)sBuf;
        for (int i = tid; i < DS_; i += 256) sWb[i] = wbv[b * DS_ + i];
        __syncthreads();
        float cbb = cb[b];
        for (int i = 0; i < 8; ++i) {
            int row = w * 8 + i;
            const float* pr  = p   + (size_t)(b * N_ + n0 + row) * DS_;
            float*       orw = out + (size_t)(b * N_ + n0 + row) * 1792;
            float s = 0.f, s2 = 0.f, dw = 0.f;
            for (int t = 0; t < 32; ++t) {
                float v = pr[lane + 32 * t];
                orw[lane + 32 * t] = v;
                s += v; s2 += v * v;
                dw = fmaf(v, sWb[lane + 32 * t], dw);
            }
            for (int o = 16; o > 0; o >>= 1) {
                s  += __shfl_xor(s,  o, 32);
                s2 += __shfl_xor(s2, o, 32);
                dw += __shfl_xor(dw, o, 32);
            }
            if (lane == 0) {
                float mu  = s * (1.f / DS_);
                float var = s2 * (1.f / DS_) - mu * mu;
                sMu[row]   = mu;
                sRstd[row] = rsqrtf(var + 1e-5f);
                sBeta[row] = 1.f / (1.f + __expf(-(dw + cbb)));
            }
        }
        __syncthreads();
    }

    const float scale = 0.10206207262f;  // 96^-0.5
    const int wr = w >> 1, wc = w & 1;
    const int rowA = tid >> 2, gA = tid & 3;             // A-stage mapping
    const int lh16A = (gA & 1) ? 16 : 0, e0A = (gA >> 1) * 8;
    const float muA = sMu[rowA], rsA = sRstd[rowA];
    const size_t prowA = (size_t)(b * N_ + n0 + rowA) * DS_;
    const unsigned int sAslot =
        ((((rowA >> 4) * 32) + (rowA & 15) + lh16A) * 16 + e0A) * 2;  // byte off

    for (int pass = 0; pass < 2; ++pass) {
        const unsigned short* WqfP = Wqf + (size_t)(pass * 32) * 12288;

        v8f acc[12];
#pragma unroll
        for (int t = 0; t < 12; ++t)
#pragma unroll
            for (int r = 0; r < 8; ++r) acc[t][r] = 0.f;

        // ---- q(64x384) = LN(p) @ Wq slice; TDM double-buffered over kc ----
        // stage A chunk `ck` into buffer `buf`: 8 contiguous K per thread
        auto stageA = [&](int ck, int buf) {
            const float* src = p + prowA + ck * 32 + 8 * gA;
            float4 va = *(const float4*)src;
            float4 vb = *(const float4*)(src + 4);
            float4 ga4 = *(const float4*)(gq + ck * 32 + 8 * gA);
            float4 gb4 = *(const float4*)(gq + ck * 32 + 8 * gA + 4);
            float4 ba4 = *(const float4*)(bq + ck * 32 + 8 * gA);
            float4 bb4 = *(const float4*)(bq + ck * 32 + 8 * gA + 4);
            unsigned int w0 = (unsigned)f2bf((va.x - muA) * rsA * ga4.x + ba4.x) |
                              ((unsigned)f2bf((va.y - muA) * rsA * ga4.y + ba4.y) << 16);
            unsigned int w1 = (unsigned)f2bf((va.z - muA) * rsA * ga4.z + ba4.z) |
                              ((unsigned)f2bf((va.w - muA) * rsA * ga4.w + ba4.w) << 16);
            unsigned int w2 = (unsigned)f2bf((vb.x - muA) * rsA * gb4.x + bb4.x) |
                              ((unsigned)f2bf((vb.y - muA) * rsA * gb4.y + bb4.y) << 16);
            unsigned int w3 = (unsigned)f2bf((vb.z - muA) * rsA * gb4.z + bb4.z) |
                              ((unsigned)f2bf((vb.w - muA) * rsA * gb4.w + bb4.w) << 16);
            uint4 pk = {w0, w1, w2, w3};
            *(uint4*)(sBuf + buf * 28672 + sAslot) = pk;    // one ds_store_b128
        };

        __syncthreads();
        stageA(0, 0);
        if (w == 0) {
            tdm_load_b24k(WqfP, (unsigned int)(size_t)(sBuf + 4096));
            __builtin_amdgcn_s_wait_tensorcnt(0);
        }
        __syncthreads();

        int cur = 0;
        for (int ck = 0; ck < 32; ++ck) {
            if (ck + 1 < 32) {
                stageA(ck + 1, cur ^ 1);
                if (w == 0)
                    tdm_load_b24k(WqfP + (size_t)(ck + 1) * 12288,
                                  (unsigned int)(size_t)(sBuf + (cur ^ 1) * 28672 + 4096));
            }
            const v16bf* sAv = (const v16bf*)(sBuf + cur * 28672);
            const v16bf* sBv = (const v16bf*)(sBuf + cur * 28672 + 4096);
            v16bf af = sAv[wr * 32 + lane];                 // 2x ds_load_b128
#pragma unroll
            for (int t = 0; t < 12; ++t) {
                v16bf bf = sBv[(12 * wc + t) * 32 + lane];
                acc[t] = __builtin_amdgcn_wmma_f32_16x16x32_bf16(
                    false, af, false, bf, (short)0, acc[t], false, false);
            }
            if (w == 0) __builtin_amdgcn_s_wait_tensorcnt(0);
            __syncthreads();
            cur ^= 1;
        }

        // ---- attention for the 4 heads living in this column pass ----
        for (int hh = 0; hh < 4; ++hh) {
            const int j = pass * 4 + hh;
            unsigned short* sQh = (unsigned short*)sBuf;            // [tr4][ks3][32][16]
            unsigned short* sKh = (unsigned short*)(sBuf + 12288);  // [tc4][ks3][32][16]
            unsigned short* sVh = (unsigned short*)(sBuf + 24576);  // [ks2][tc6][32][16]
            float*          sAl = (float*)(sBuf + 36864);           // 64x64 f32
            unsigned short* sGm = (unsigned short*)sBuf;            // [tr4][ks2][32][16]

            __syncthreads();
            {   // stage K_h / V_h: pure b128 copies (already frag-ordered)
                const uint4* Khv = (const uint4*)(Kbf + (size_t)(b * H_ + j) * 6144);
                const uint4* Vhv = (const uint4*)(Vbf + (size_t)(b * H_ + j) * 6144);
                uint4* dK = (uint4*)sKh;
                uint4* dV = (uint4*)sVh;
                for (int i = tid; i < 768; i += 256) { dK[i] = Khv[i]; dV[i] = Vhv[i]; }
            }
            if (wc == (hh >> 1)) {                  // spill q_h tile into A-frag order
                int tb = 6 * (hh & 1);
#pragma unroll
                for (int tt = 0; tt < 6; ++tt)
#pragma unroll
                    for (int r = 0; r < 8; ++r) {
                        int mrow = r + (lhi ? 8 : 0);
                        int colh = 16 * tt + m;
                        int gcol = pass * 384 + 192 * wc + colh;
                        int ks = colh >> 5, kk = colh & 31, lh16, e;
                        fragpos(kk, lh16, e);
                        sQh[((wr * 3 + ks) * 32 + mrow + lh16) * 16 + e] =
                            f2bf(acc[tb + tt][r] + bql[gcol]);
                    }
            }
            __syncthreads();

            // alpha(64x64) = q_h(64x96) @ K_h^T
            const v16bf* sQv = (const v16bf*)sQh;
            const v16bf* sKv = (const v16bf*)sKh;
#pragma unroll
            for (int i = 0; i < 2; ++i) {
                int t = 2 * w + i, tr = t >> 2, tc = t & 3;
                v8f aacc;
#pragma unroll
                for (int r = 0; r < 8; ++r) aacc[r] = 0.f;
#pragma unroll
                for (int ks = 0; ks < 3; ++ks) {
                    v16bf qa = sQv[(tr * 3 + ks) * 32 + lane];
                    v16bf kb = sKv[(tc * 3 + ks) * 32 + lane];
                    aacc = __builtin_amdgcn_wmma_f32_16x16x32_bf16(
                        false, qa, false, kb, (short)0, aacc, false, false);
                }
#pragma unroll
                for (int r = 0; r < 8; ++r)
                    sAl[(16 * tr + r + (lhi ? 8 : 0)) * 64 + 16 * tc + m] = aacc[r] * scale;
            }
            __syncthreads();

            // softmax over l per row -> gamma (A-frag order)
            if (tid < 64) {
                float mx = -1e30f;
                for (int q = 0; q < 64; ++q) mx = fmaxf(mx, sAl[tid * 64 + q]);
                float sum = 0.f;
                for (int q = 0; q < 64; ++q) sum += __expf(sAl[tid * 64 + q] - mx);
                float inv = 1.f / sum;
                int tr = tid >> 4, mrow = tid & 15;
                for (int q = 0; q < 64; ++q) {
                    int ks = q >> 5, kk = q & 31, lh16, e;
                    fragpos(kk, lh16, e);
                    sGm[((tr * 2 + ks) * 32 + mrow + lh16) * 16 + e] =
                        f2bf(__expf(sAl[tid * 64 + q] - mx) * inv);
                }
            }
            __syncthreads();

            // C_h(64x96) = gamma(64x64) @ V_h(64x96); row-scale by beta
            const v16bf* sGv = (const v16bf*)sGm;
            const v16bf* sVv = (const v16bf*)sVh;
#pragma unroll
            for (int i = 0; i < 3; ++i) {
                int t = 3 * w + i, tr = t / 6, tc = t - tr * 6;
                v8f cacc;
#pragma unroll
                for (int r = 0; r < 8; ++r) cacc[r] = 0.f;
#pragma unroll
                for (int ks = 0; ks < 2; ++ks) {
                    v16bf ga = sGv[(tr * 2 + ks) * 32 + lane];
                    v16bf vb = sVv[(ks * 6 + tc) * 32 + lane];
                    cacc = __builtin_amdgcn_wmma_f32_16x16x32_bf16(
                        false, ga, false, vb, (short)0, cacc, false, false);
                }
#pragma unroll
                for (int r = 0; r < 8; ++r) {
                    int row = 16 * tr + r + (lhi ? 8 : 0);
                    int col = j * HD_ + 16 * tc + m;
                    out[(size_t)(b * N_ + n0 + row) * 1792 + 1024 + col] =
                        cacc[r] * sBeta[row];
                }
            }
        }
    }
}

// ============================================================================
extern "C" void kernel_launch(void* const* d_in, const int* in_sizes, int n_in,
                              void* d_out, int out_size, void* d_ws, size_t ws_size,
                              hipStream_t stream) {
    const float* p   = (const float*)d_in[0];
    const float* u   = (const float*)d_in[1];
    const float* gq  = (const float*)d_in[2];
    const float* bq  = (const float*)d_in[3];
    const float* gk  = (const float*)d_in[4];
    const float* bk  = (const float*)d_in[5];
    const float* gv  = (const float*)d_in[6];
    const float* bv  = (const float*)d_in[7];
    const float* Wq  = (const float*)d_in[8];
    const float* bql = (const float*)d_in[9];
    const float* Wk  = (const float*)d_in[10];
    const float* bkl = (const float*)d_in[11];
    const float* Wv  = (const float*)d_in[12];
    const float* bvl = (const float*)d_in[13];
    const float* Wqb = (const float*)d_in[14];
    const float* bqb = (const float*)d_in[15];
    const float* Wkb = (const float*)d_in[16];
    const float* bkb = (const float*)d_in[17];
    float* out = (float*)d_out;

    unsigned char* ws = (unsigned char*)d_ws;
    unsigned short* Kbf = (unsigned short*)(ws);                 //  768 KB
    unsigned short* Vbf = (unsigned short*)(ws + 786432);        //  768 KB
    float* kbeta = (float*)(ws + 1572864);                       //   32 KB
    float* wbv   = (float*)(ws + 1605632);                       //   32 KB
    float* cbuf  = (float*)(ws + 1638400);                       //   32 B
    unsigned short* Wqf = (unsigned short*)(ws + 1638432);       //  1.5 MB

    wqprep_kernel<<<dim3(64), dim3(256), 0, stream>>>(Wq, Wqf);
    kv_kernel<<<dim3(8), dim3(256), 0, stream>>>(u, gk, bk, gv, bv, Wk, bkl, Wv, bvl,
                                                 Kbf, Vbf);
    kbeta_kernel<<<dim3(16), dim3(256), 0, stream>>>(u, Wkb, bkb, kbeta);
    wb_kernel<<<dim3(1032), dim3(256), 0, stream>>>(Wqb, bqb, kbeta, wbv, cbuf);
    fused_kernel<<<dim3(256), dim3(256), 0, stream>>>(p, gq, bq, Wqf, bql, Kbf, Vbf,
                                                      wbv, cbuf, out);
}
